// LowRankSparseCoder_24919400251818
// MI455X (gfx1250) — matrix operand order repaired and checked by reading, not compiled, not run on hardware
//
#include <hip/hip_runtime.h>
#include <hip/hip_bf16.h>

// Problem sizes (fixed by the reference)
#define N_TOK 4096
#define D_DIM 1024
#define M_LAT 8192
#define R_RANK 64
#define K_TOP 32

typedef __attribute__((ext_vector_type(16))) __bf16 v16bf;
typedef __attribute__((ext_vector_type(8)))  __bf16 v8bf;   // 16 bytes
typedef __attribute__((ext_vector_type(2)))  __bf16 v2bf;
typedef __attribute__((ext_vector_type(8)))  float  v8f;

#if defined(__has_builtin)
#if __has_builtin(__builtin_amdgcn_cvt_pk_bf16_f32)
#define HAVE_PK_BF16 1
#endif
#endif

__device__ __forceinline__ __bf16 f2bf(float f) {
  union { float f; unsigned int u; } a; a.f = f;
  unsigned int r = (a.u + 0x7FFFu + ((a.u >> 16) & 1u)) >> 16;  // RNE
  unsigned short s = (unsigned short)r;
  __bf16 b;
  __builtin_memcpy(&b, &s, sizeof(b));
  return b;
}

__device__ __forceinline__ v2bf cvt2(float a, float b) {
#ifdef HAVE_PK_BF16
  return __builtin_amdgcn_cvt_pk_bf16_f32(a, b);
#else
  v2bf r; r[0] = f2bf(a); r[1] = f2bf(b); return r;
#endif
}

__device__ __forceinline__ v8bf cvt8(float4 u, float4 v) {
  v2bf a = cvt2(u.x, u.y), b = cvt2(u.z, u.w);
  v2bf c = cvt2(v.x, v.y), d = cvt2(v.z, v.w);
  v8bf r;
  r[0] = a[0]; r[1] = a[1]; r[2] = b[0]; r[3] = b[1];
  r[4] = c[0]; r[5] = c[1]; r[6] = d[0]; r[7] = d[1];
  return r;
}

// Build a WMMA fragment from two 16-byte bf16 chunks
__device__ __forceinline__ v16bf load_frag(const __bf16* __restrict__ p0,
                                           const __bf16* __restrict__ p1) {
  v8bf lo = *(const v8bf*)p0;
  v8bf hi = *(const v8bf*)p1;
  return __builtin_shufflevector(lo, hi, 0, 1, 2, 3, 4, 5, 6, 7,
                                 8, 9, 10, 11, 12, 13, 14, 15);
}

#define WMMA_BF16(A_, B_, C_) \
  __builtin_amdgcn_wmma_f32_16x16x32_bf16(false, (A_), false, (B_), (short)0, (C_), false, false)

// ---------------------------------------------------------------------------
// Kernel 0: zero the reduction accumulators (colsum[D] + scal[2])
// ---------------------------------------------------------------------------
__global__ void k_init(float* __restrict__ p, int n) {
  int i = blockIdx.x * blockDim.x + threadIdx.x;
  if (i < n) p[i] = 0.0f;
}

// ---------------------------------------------------------------------------
// Generic f32 -> bf16 copy, 8 elements per thread (for A and B matrices)
// ---------------------------------------------------------------------------
__global__ __launch_bounds__(256) void k_cvt(const float* __restrict__ src,
                                             __bf16* __restrict__ dst, int n8) {
  int i = blockIdx.x * blockDim.x + threadIdx.x;
  if (i >= n8) return;
  const float* p = src + (size_t)i * 8;
  float4 u = *(const float4*)p;
  float4 v = *(const float4*)(p + 4);
  *(v8bf*)(dst + (size_t)i * 8) = cvt8(u, v);
}

// ---------------------------------------------------------------------------
// xc_bf = bf16(x - b_dec), 8 elements per thread
// ---------------------------------------------------------------------------
__global__ __launch_bounds__(256) void k_prep_x(const float* __restrict__ x,
                                                const float* __restrict__ b_dec,
                                                __bf16* __restrict__ xc_bf) {
  int i = blockIdx.x * blockDim.x + threadIdx.x;   // 8-element group id
  const size_t base = (size_t)i * 8;
  const int d = (int)(base & (D_DIM - 1));
  float4 u = *(const float4*)(x + base);
  float4 v = *(const float4*)(x + base + 4);
  float4 a = *(const float4*)(b_dec + d);
  float4 b = *(const float4*)(b_dec + d + 4);
  u.x -= a.x; u.y -= a.y; u.z -= a.z; u.w -= a.w;
  v.x -= b.x; v.y -= b.y; v.z -= b.z; v.w -= b.w;
  *(v8bf*)(xc_bf + base) = cvt8(u, v);
}

// ---------------------------------------------------------------------------
// Kernel 1: T_bf[N,R] = bf16( xc_bf @ B_bf^T )   (pure loads + WMMA)
// One wave per 16x16 tile; 1024 waves (256 n-tiles x 4 r-tiles); 32 k-steps.
// ---------------------------------------------------------------------------
__global__ __launch_bounds__(256) void k1_lowrank(const __bf16* __restrict__ xc_bf,
                                                  const __bf16* __restrict__ B_bf,
                                                  __bf16* __restrict__ T_bf) {
  const int lane = threadIdx.x & 31;
  const int wave = blockIdx.x * (blockDim.x >> 5) + (threadIdx.x >> 5);
  const int n0 = (wave >> 2) * 16;
  const int r0 = (wave & 3) * 16;
  const int hi = lane >> 4;
  const int l15 = lane & 15;
  const int offA = hi * 8;    // A-frag K offset (0 or 8)
  const int offB = hi * 8;    // B-frag K offset (0 or 8)
  const __bf16* xr = xc_bf + (size_t)(n0 + l15) * D_DIM;
  const __bf16* br = B_bf + (size_t)(r0 + l15) * D_DIM;

  v8f acc = {};
  for (int kk = 0; kk < D_DIM / 32; ++kk) {
    const int d0 = kk * 32;
    v16bf af = load_frag(xr + d0 + offA, xr + d0 + 16 + offA);
    v16bf bf_ = load_frag(br + d0 + offB, br + d0 + offB + 16);
    acc = WMMA_BF16(af, bf_, acc);
  }
  #pragma unroll
  for (int i = 0; i < 8; ++i)
    T_bf[(size_t)(n0 + i + hi * 8) * R_RANK + r0 + l15] = f2bf(acc[i]);
}

// ---------------------------------------------------------------------------
// Kernel 2: pre_acts[N,M] = ReLU(T @ A^T + bias)
// Each wave holds FOUR T-fragments (4 n-tiles) and reuses every A_bf
// B-fragment across them: 8 WMMAs per B-fragment load.
// grid (64 n-groups, 16); 8 waves/block; each wave: 4 m-tiles.
// ---------------------------------------------------------------------------
__global__ __launch_bounds__(256) void k2_expand(const __bf16* __restrict__ T_bf,
                                                 const __bf16* __restrict__ A_bf,
                                                 const float* __restrict__ bias,
                                                 float* __restrict__ pre) {
  const int lane = threadIdx.x & 31;
  const int hi = lane >> 4, l15 = lane & 15;
  const int n0 = blockIdx.x * 64;                                        // 4 n-tiles
  const int strip = blockIdx.y * (blockDim.x >> 5) + (threadIdx.x >> 5); // 0..127
  const int offA = hi * 8;
  const int offB = hi * 8;

  // 4 T-fragments (A operand), each split into 2 K-steps
  v16bf t0[4], t1[4];
  #pragma unroll
  for (int nt = 0; nt < 4; ++nt) {
    const __bf16* tp = T_bf + (size_t)(n0 + nt * 16 + l15) * R_RANK;
    t0[nt] = load_frag(tp + offA,      tp + 16 + offA);   // K 0..31
    t1[nt] = load_frag(tp + 32 + offA, tp + 48 + offA);   // K 32..63
  }

  for (int mt = 0; mt < 4; ++mt) {
    const int m0 = (strip * 4 + mt) * 16;
    const __bf16* ap = A_bf + (size_t)(m0 + l15) * R_RANK + offB;
    v16bf b0 = load_frag(ap, ap + 16);           // K 0..31
    v16bf b1 = load_frag(ap + 32, ap + 48);      // K 32..63
    const float bv = bias[m0 + l15];

    v8f acc[4];
    #pragma unroll
    for (int nt = 0; nt < 4; ++nt) {
      acc[nt] = (v8f){};
      acc[nt] = WMMA_BF16(t0[nt], b0, acc[nt]);
      acc[nt] = WMMA_BF16(t1[nt], b1, acc[nt]);
    }
    #pragma unroll
    for (int nt = 0; nt < 4; ++nt) {
      #pragma unroll
      for (int i = 0; i < 8; ++i) {
        float v = acc[nt][i] + bv;
        v = v > 0.0f ? v : 0.0f;                 // ReLU
        pre[(size_t)(n0 + nt * 16 + i + hi * 8) * M_LAT + m0 + l15] = v;
      }
    }
  }
}

// ---------------------------------------------------------------------------
// Kernel 3: column sums of x (for total_variance)
// ---------------------------------------------------------------------------
__global__ __launch_bounds__(256) void k_colsum(const float* __restrict__ x,
                                                float* __restrict__ colsum) {
  const int t = threadIdx.x;
  const int r0 = blockIdx.x * 16;
  float s0 = 0.f, s1 = 0.f, s2 = 0.f, s3 = 0.f;
  for (int r = 0; r < 16; ++r) {
    float4 v = *(const float4*)(x + (size_t)(r0 + r) * D_DIM + t * 4);
    s0 += v.x; s1 += v.y; s2 += v.z; s3 += v.w;
  }
  atomicAdd(&colsum[t * 4 + 0], s0);
  atomicAdd(&colsum[t * 4 + 1], s1);
  atomicAdd(&colsum[t * 4 + 2], s2);
  atomicAdd(&colsum[t * 4 + 3], s3);
}

// ---------------------------------------------------------------------------
// Kernel 4: per-row top-32 (descending, ties -> lowest index)
// ---------------------------------------------------------------------------
__global__ __launch_bounds__(256) void k_topk(const float* __restrict__ pre,
                                              float* __restrict__ acts,
                                              int* __restrict__ idxo) {
  __shared__ float sv[256];
  __shared__ int   si[256];
  const int row = blockIdx.x;
  const int t = threadIdx.x;
  const float* pr = pre + (size_t)row * M_LAT;
  float v[32];
  #pragma unroll
  for (int j = 0; j < 32; ++j) v[j] = pr[t + j * 256];  // coalesced, m = t + 256j

  for (int r = 0; r < K_TOP; ++r) {
    float bv = -__builtin_inff(); int bj = 0;
    #pragma unroll
    for (int j = 0; j < 32; ++j)
      if (v[j] > bv) { bv = v[j]; bj = j; }
    sv[t] = bv; si[t] = t + bj * 256;
    __syncthreads();
    for (int s = 128; s > 0; s >>= 1) {
      if (t < s) {
        float ov = sv[t + s]; int oi = si[t + s];
        if (ov > sv[t] || (ov == sv[t] && oi < si[t])) { sv[t] = ov; si[t] = oi; }
      }
      __syncthreads();
    }
    const float wv = sv[0];
    const int wm = si[0];
    if (t == 0) { acts[(size_t)row * K_TOP + r] = wv; idxo[(size_t)row * K_TOP + r] = wm; }
    const int wj = wm >> 8;
    #pragma unroll
    for (int j = 0; j < 32; ++j)
      if (((wm & 255) == t) && (j == wj)) v[j] = -__builtin_inff();
    __syncthreads();
  }
}

// ---------------------------------------------------------------------------
// Kernel 5: sae_out[row,:] = sum_k act_k * W_dec[idx_k,:] + b_dec
// ---------------------------------------------------------------------------
__global__ __launch_bounds__(256) void k_decode(const float* __restrict__ acts,
                                                const int* __restrict__ idx,
                                                const float* __restrict__ W_dec,
                                                const float* __restrict__ b_dec,
                                                float* __restrict__ out) {
  __shared__ float sa[K_TOP];
  __shared__ int   sx[K_TOP];
  const int row = blockIdx.x, t = threadIdx.x;
  if (t < K_TOP) { sa[t] = acts[(size_t)row * K_TOP + t]; sx[t] = idx[(size_t)row * K_TOP + t]; }
  __syncthreads();
  float acc[4];
  #pragma unroll
  for (int j = 0; j < 4; ++j) acc[j] = b_dec[t + j * 256];
  for (int k = 0; k < K_TOP; ++k) {
    const float w = sa[k];
    const float* wr = W_dec + (size_t)sx[k] * D_DIM;
    #pragma unroll
    for (int j = 0; j < 4; ++j) acc[j] += w * wr[t + j * 256];
  }
  #pragma unroll
  for (int j = 0; j < 4; ++j) out[(size_t)row * D_DIM + t + j * 256] = acc[j];
}

// ---------------------------------------------------------------------------
// Kernel 6: accumulate total_variance and l2_loss
// ---------------------------------------------------------------------------
__global__ __launch_bounds__(256) void k_loss(const float* __restrict__ x,
                                              const float* __restrict__ sae,
                                              const float* __restrict__ colsum,
                                              float* __restrict__ scal) {
  __shared__ float stv[256], sl2[256];
  float tv = 0.f, l2 = 0.f;
  const float invN = 1.0f / (float)N_TOK;
  const size_t total = (size_t)N_TOK * D_DIM;
  for (size_t i = (size_t)blockIdx.x * blockDim.x + threadIdx.x; i < total;
       i += (size_t)gridDim.x * blockDim.x) {
    const int d = (int)(i & (D_DIM - 1));
    const float xv = x[i];
    const float e1 = xv - colsum[d] * invN;
    const float e2 = xv - sae[i];
    tv += e1 * e1;
    l2 += e2 * e2;
  }
  stv[threadIdx.x] = tv; sl2[threadIdx.x] = l2;
  __syncthreads();
  for (int s = 128; s > 0; s >>= 1) {
    if (threadIdx.x < s) {
      stv[threadIdx.x] += stv[threadIdx.x + s];
      sl2[threadIdx.x] += sl2[threadIdx.x + s];
    }
    __syncthreads();
  }
  if (threadIdx.x == 0) { atomicAdd(&scal[0], stv[0]); atomicAdd(&scal[1], sl2[0]); }
}

__global__ void k_final(const float* __restrict__ scal, float* __restrict__ fvu_out) {
  fvu_out[0] = scal[1] / scal[0];  // fvu = l2 / total_variance
  fvu_out[1] = 0.0f;               // auxk_loss
  fvu_out[2] = 0.0f;               // multi_topk_fvu
}

// ---------------------------------------------------------------------------
extern "C" void kernel_launch(void* const* d_in, const int* in_sizes, int n_in,
                              void* d_out, int out_size, void* d_ws, size_t ws_size,
                              hipStream_t stream) {
  const float* x     = (const float*)d_in[0];
  const float* A     = (const float*)d_in[1];
  const float* Bmat  = (const float*)d_in[2];
  const float* bias  = (const float*)d_in[3];
  const float* W_dec = (const float*)d_in[4];
  const float* b_dec = (const float*)d_in[5];
  // d_in[6] is k == 32 (compile-time K_TOP)

  // Workspace layout (bytes):
  // pre (128MB, L2-resident) | xc_bf (8MB) | A_bf (1MB) | B_bf (128KB) |
  // T_bf (512KB) | colsum (4KB) | scal (8B)
  char* ws = (char*)d_ws;
  float*  pre    = (float*)ws;                                   ws += (size_t)N_TOK * M_LAT * 4;
  __bf16* xc_bf  = (__bf16*)ws;                                  ws += (size_t)N_TOK * D_DIM * 2;
  __bf16* A_bf   = (__bf16*)ws;                                  ws += (size_t)M_LAT * R_RANK * 2;
  __bf16* B_bf   = (__bf16*)ws;                                  ws += (size_t)R_RANK * D_DIM * 2;
  __bf16* T_bf   = (__bf16*)ws;                                  ws += (size_t)N_TOK * R_RANK * 2;
  float*  colsum = (float*)ws;                                   ws += (size_t)D_DIM * 4;
  float*  scal   = colsum + D_DIM;  // 2 floats right after colsum

  // Output layout: sae_out | top_acts | top_indices (int bits) | fvu,auxk,multi
  float* out      = (float*)d_out;
  float* sae_out  = out;
  float* acts_out = out + (size_t)N_TOK * D_DIM;
  int*   idx_out  = (int*)(out + (size_t)N_TOK * D_DIM + (size_t)N_TOK * K_TOP);
  float* fvu_out  = out + (size_t)N_TOK * D_DIM + 2 * (size_t)N_TOK * K_TOP;

  k_init<<<(D_DIM + 2 + 255) / 256, 256, 0, stream>>>(colsum, D_DIM + 2);
  k_cvt<<<(M_LAT * R_RANK / 8 + 255) / 256, 256, 0, stream>>>(A, A_bf, M_LAT * R_RANK / 8);
  k_cvt<<<(R_RANK * D_DIM / 8 + 255) / 256, 256, 0, stream>>>(Bmat, B_bf, R_RANK * D_DIM / 8);
  k_prep_x<<<N_TOK * D_DIM / 8 / 256, 256, 0, stream>>>(x, b_dec, xc_bf);
  k1_lowrank<<<128, 256, 0, stream>>>(xc_bf, B_bf, T_bf);
  k2_expand<<<dim3(64, 16), 256, 0, stream>>>(T_bf, A_bf, bias, pre);
  k_colsum<<<N_TOK / 16, 256, 0, stream>>>(x, colsum);
  k_topk<<<N_TOK, 256, 0, stream>>>(pre, acts_out, idx_out);
  k_decode<<<N_TOK, 256, 0, stream>>>(acts_out, idx_out, W_dec, b_dec, sae_out);
  k_loss<<<512, 256, 0, stream>>>(x, sae_out, colsum, scal);
  k_final<<<1, 1, 0, stream>>>(scal, fvu_out);
}
